// CorefGRU_32074815766703
// MI455X (gfx1250) — compile-verified, matrix operand order checked by reading, not run on hardware
//
#include <hip/hip_runtime.h>
#include <hip/hip_bf16.h>

// CorefGRU for MI455X (gfx1250, wave32).
// Strategy: bf16 WMMA (v_wmma_f32_16x16x32_bf16, f32 accum) for all GEMMs.
// - xz/xr input projections: massively parallel wave-per-tile GEMM.
// - recurrence: one persistent kernel, grid-wide barrier between the
//   h@Uz^T / h@Ur^T phase and the dependent (r*h)@Uz^T phase. Weights stay
//   resident in the 192MB L2 (2MB each in bf16), so the scan is latency-
//   bound, not bandwidth-bound -> minimize launches + per-step work.

#define T_STEPS 512
#define BATCH   64
#define DIM     1024
#define BD      (BATCH * DIM)   // 65536

typedef __attribute__((ext_vector_type(16))) __bf16 v16bf;
typedef __attribute__((ext_vector_type(8)))  float  v8f;

// ---- WMMA helpers -----------------------------------------------------------

__device__ __forceinline__ v8f wmma_bf16(v16bf a, v16bf b, v8f c) {
  // 8 args: (neg_a, A, neg_b, B, c_mod, C, reuse_a, reuse_b)
  return __builtin_amdgcn_wmma_f32_16x16x32_bf16(false, a, false, b,
                                                 (short)0, c, false, false);
}

// 16-bit A-matrix fragment layout (ISA 7.12.2): lane holds one M row (or one
// N column for B); per lane, elements 0..7 are K = k0..k0+7 and elements
// 8..15 are K = k0+16..k0+23 (lanes 16..31 add +8). Caller passes
// p = row_base + k + (lane>>4)*8, so we read two contiguous 16B groups.
__device__ __forceinline__ v16bf frag_ld(const __hip_bfloat16* p_) {
  const __bf16* p = reinterpret_cast<const __bf16*>(p_);
  v16bf f;
#pragma unroll
  for (int i = 0; i < 8; ++i) {
    f[i]     = p[i];
    f[i + 8] = p[16 + i];
  }
  return f;
}

// One 16x16 C tile of  X[m0.., :] @ W[n0.., :]^T  over K in [kbeg, kend).
// X rows are M (stride DIM), W rows are N (stride DIM) -> both K-contiguous.
__device__ __forceinline__ v8f tile_gemm(const __hip_bfloat16* X,
                                         const __hip_bfloat16* W,
                                         int m0, int n0, int kbeg, int kend,
                                         int lane) {
  v8f acc = {};
  const __hip_bfloat16* arow = X + (size_t)(m0 + (lane & 15)) * DIM + ((lane >> 4) * 8);
  const __hip_bfloat16* brow = W + (size_t)(n0 + (lane & 15)) * DIM + ((lane >> 4) * 8);
  for (int k = kbeg; k < kend; k += 32) {
    __builtin_prefetch(brow + k + 256, 0, 0);  // global_prefetch_b8 on weights
    v16bf a = frag_ld(arow + k);
    v16bf b = frag_ld(brow + k);
    acc = wmma_bf16(a, b, acc);
  }
  return acc;
}

__device__ __forceinline__ float sigmoidf_(float x) {
  return 1.0f / (1.0f + __expf(-x));
}

// ---- grid-wide barrier (persistent kernel) ----------------------------------

struct GridBar { unsigned count; unsigned gen; };

__device__ __forceinline__ void grid_barrier(GridBar* bar) {
  __threadfence();
  __syncthreads();
  if (threadIdx.x == 0) {
    unsigned target = atomicAdd(&bar->gen, 0u) + 1u;   // capture before arrive
    unsigned prev   = atomicAdd(&bar->count, 1u);
    if (prev == gridDim.x - 1) {
      bar->count = 0u;
      __threadfence();
      atomicAdd(&bar->gen, 1u);
    } else {
      while (atomicAdd(&bar->gen, 0u) < target)
        __builtin_amdgcn_s_sleep(2);
    }
  }
  __syncthreads();
  __threadfence();
}

// ---- prep kernels -----------------------------------------------------------

__global__ void cvt_f32_bf16(const float* __restrict__ s,
                             __hip_bfloat16* __restrict__ d, int n) {
  int i = blockIdx.x * blockDim.x + threadIdx.x;
  if (i < n) d[i] = __float2bfloat16(s[i]);
}

__global__ void init_state(float* __restrict__ zerobuf,
                           __hip_bfloat16* __restrict__ hbf,
                           unsigned* __restrict__ bar) {
  int i = blockIdx.x * blockDim.x + threadIdx.x;
  if (i < BD) {
    zerobuf[i] = 0.0f;
    hbf[i]     = __float2bfloat16(0.0f);
  }
  if (i < 2) bar[i] = 0u;
}

// ---- input projection: xz = inp@Wz^T + bz ; xr = inp@Wr^T + br --------------
// One wave per 16x16 output tile. Rows span T*B = 32768 (2048 M-tiles),
// cols span DIM (64 N-tiles); sel picks the z or r projection.

__global__ __launch_bounds__(256) void xproj_gemm(
    const __hip_bfloat16* __restrict__ inp_bf,
    const __hip_bfloat16* __restrict__ Wz_bf,
    const __hip_bfloat16* __restrict__ Wr_bf,
    const float* __restrict__ bz, const float* __restrict__ br,
    __hip_bfloat16* __restrict__ xz, __hip_bfloat16* __restrict__ xr) {
  const int lane = threadIdx.x & 31;
  const int wid  = threadIdx.x >> 5;
  const int w    = blockIdx.x * 8 + wid;

  const int tilesPer = (T_STEPS * BATCH / 16) * (DIM / 16);  // 131072 = 2^17
  const int sel  = w >> 17;
  const int tile = w & (tilesPer - 1);
  const int m0   = (tile >> 6) * 16;
  const int n0   = (tile & 63) * 16;

  const __hip_bfloat16* W = sel ? Wr_bf : Wz_bf;
  v8f acc = tile_gemm(inp_bf, W, m0, n0, 0, DIM, lane);

  const float* bias = sel ? br : bz;
  __hip_bfloat16* dst = sel ? xr : xz;
  const int n  = lane & 15;
  const int mb = (lane >> 4) * 8;
  const float bv = bias[n0 + n];
#pragma unroll
  for (int v = 0; v < 8; ++v) {
    size_t idx = (size_t)(m0 + mb + v) * DIM + (n0 + n);
    dst[idx] = __float2bfloat16(acc[v] + bv);
  }
}

// ---- persistent recurrence --------------------------------------------------
// 64 blocks x 8 waves = 512 waves.
// Phase 1: waves 0..255 -> z tiles (h@Uz^T), waves 256..511 -> r tiles
//          (h@Ur^T, then store r*h in bf16).  [grid barrier]
// Phase 2: 256 tiles of (r*h)@Uz^T split-K over 512 waves; partner waves
//          (same block) reduce via LDS; finalize h_new.  [grid barrier]

__global__ __launch_bounds__(256) void coref_gru_scan(
    const __hip_bfloat16* __restrict__ xz,
    const __hip_bfloat16* __restrict__ xr,
    const __hip_bfloat16* __restrict__ Uz_bf,
    const __hip_bfloat16* __restrict__ Ur_bf,
    const float* __restrict__ zerobuf,
    __hip_bfloat16* __restrict__ hbf,     // [B, D] bf16 copy of h (WMMA operand)
    float* __restrict__ zbuf,             // [B, D] z gate
    __hip_bfloat16* __restrict__ rhbf,    // [B, D] bf16 r*h
    GridBar* __restrict__ bar,
    float* __restrict__ out) {            // [T, B, D] doubles as h history
  __shared__ float red[8 * 32 * 8];       // 8KB split-K partials

  const int lane = threadIdx.x & 31;
  const int wid  = threadIdx.x >> 5;
  const int w    = blockIdx.x * 8 + wid;  // 0..511
  const int n    = lane & 15;
  const int mb   = (lane >> 4) * 8;

  for (int t = 0; t < T_STEPS; ++t) {
    const float* hprev = (t == 0) ? zerobuf : out + (size_t)(t - 1) * BD;
    const __hip_bfloat16* xz_t = xz + (size_t)t * BD;
    const __hip_bfloat16* xr_t = xr + (size_t)t * BD;

    // ---------------- phase 1: gates ----------------
    {
      const int gem  = w >> 8;          // 0 -> z gate, 1 -> r gate
      const int tile = w & 255;
      const int m0   = (tile >> 6) * 16;
      const int n0   = (tile & 63) * 16;
      const __hip_bfloat16* U = gem ? Ur_bf : Uz_bf;
      v8f acc = tile_gemm(hbf, U, m0, n0, 0, DIM, lane);

      if (gem == 0) {
#pragma unroll
        for (int v = 0; v < 8; ++v) {
          size_t idx = (size_t)(m0 + mb + v) * DIM + (n0 + n);
          float pre = __bfloat162float(xz_t[idx]) + acc[v];
          zbuf[idx] = sigmoidf_(pre);
        }
      } else {
#pragma unroll
        for (int v = 0; v < 8; ++v) {
          size_t idx = (size_t)(m0 + mb + v) * DIM + (n0 + n);
          float pre = __bfloat162float(xr_t[idx]) + acc[v];
          float r   = sigmoidf_(pre);
          rhbf[idx] = __float2bfloat16(r * hprev[idx]);
        }
      }
    }
    grid_barrier(bar);

    // ---------------- phase 2: candidate + update ----------------
    {
      const int tile = w >> 1;
      const int kh   = w & 1;           // split-K half
      const int m0   = (tile >> 6) * 16;
      const int n0   = (tile & 63) * 16;
      const int kbeg = kh * (DIM / 2);
      v8f acc = tile_gemm(rhbf, Uz_bf, m0, n0, kbeg, kbeg + DIM / 2, lane);

      float* my = &red[(wid * 32 + lane) * 8];
#pragma unroll
      for (int v = 0; v < 8; ++v) my[v] = acc[v];
      __syncthreads();

      if ((wid & 1) == 0) {
        const float* other = &red[((wid + 1) * 32 + lane) * 8];
        float* out_t = out + (size_t)t * BD;
#pragma unroll
        for (int v = 0; v < 8; ++v) {
          size_t idx = (size_t)(m0 + mb + v) * DIM + (n0 + n);
          float ap  = my[v] + other[v];
          float z   = zbuf[idx];
          float cnd = tanhf(__bfloat162float(xz_t[idx]) + ap);
          float hn  = (1.0f - z) * hprev[idx] + z * cnd;
          out_t[idx] = hn;                       // h history == output
          hbf[idx]   = __float2bfloat16(hn);     // next step's WMMA operand
        }
      }
    }
    grid_barrier(bar);
  }
}

// ---- host launch ------------------------------------------------------------

extern "C" void kernel_launch(void* const* d_in, const int* in_sizes, int n_in,
                              void* d_out, int out_size, void* d_ws, size_t ws_size,
                              hipStream_t stream) {
  const float* inp = (const float*)d_in[0];
  // d_in[1] = last_coref_idx : unused by the reference
  const float* Wz  = (const float*)d_in[2];
  const float* bz  = (const float*)d_in[3];
  const float* Uz  = (const float*)d_in[4];
  const float* Wr  = (const float*)d_in[5];
  const float* br  = (const float*)d_in[6];
  const float* Ur  = (const float*)d_in[7];
  float* out = (float*)d_out;

  char* ws = (char*)d_ws;
  size_t off = 0;
  auto take = [&](size_t bytes) -> char* {
    off = (off + 255) & ~(size_t)255;
    char* p = ws + off;
    off += bytes;
    return p;
  };

  GridBar*        bar     = (GridBar*)       take(256);
  float*          zerobuf = (float*)         take((size_t)BD * 4);
  __hip_bfloat16* hbf     = (__hip_bfloat16*)take((size_t)BD * 2);
  float*          zbuf    = (float*)         take((size_t)BD * 4);
  __hip_bfloat16* rhbf    = (__hip_bfloat16*)take((size_t)BD * 2);
  __hip_bfloat16* Wz_bf   = (__hip_bfloat16*)take((size_t)DIM * DIM * 2);
  __hip_bfloat16* Wr_bf   = (__hip_bfloat16*)take((size_t)DIM * DIM * 2);
  __hip_bfloat16* Uz_bf   = (__hip_bfloat16*)take((size_t)DIM * DIM * 2);
  __hip_bfloat16* Ur_bf   = (__hip_bfloat16*)take((size_t)DIM * DIM * 2);
  __hip_bfloat16* inp_bf  = (__hip_bfloat16*)take((size_t)T_STEPS * BD * 2);
  __hip_bfloat16* xz_bf   = (__hip_bfloat16*)take((size_t)T_STEPS * BD * 2);
  __hip_bfloat16* xr_bf   = (__hip_bfloat16*)take((size_t)T_STEPS * BD * 2);

  const int nW = DIM * DIM;              // 1M
  const int nI = T_STEPS * BD;           // 33.5M
  cvt_f32_bf16<<<(nI + 255) / 256, 256, 0, stream>>>(inp, inp_bf, nI);
  cvt_f32_bf16<<<(nW + 255) / 256, 256, 0, stream>>>(Wz, Wz_bf, nW);
  cvt_f32_bf16<<<(nW + 255) / 256, 256, 0, stream>>>(Wr, Wr_bf, nW);
  cvt_f32_bf16<<<(nW + 255) / 256, 256, 0, stream>>>(Uz, Uz_bf, nW);
  cvt_f32_bf16<<<(nW + 255) / 256, 256, 0, stream>>>(Ur, Ur_bf, nW);

  init_state<<<(BD + 255) / 256, 256, 0, stream>>>(zerobuf, hbf, (unsigned*)bar);

  // 2 projections * 131072 tiles, one wave each -> 32768 blocks of 8 waves.
  xproj_gemm<<<32768, 256, 0, stream>>>(inp_bf, Wz_bf, Wr_bf, bz, br, xz_bf, xr_bf);

  // Persistent scan: 64 co-resident blocks (trivially resident on MI455X).
  coref_gru_scan<<<64, 256, 0, stream>>>(xz_bf, xr_bf, Uz_bf, Ur_bf, zerobuf,
                                         hbf, zbuf, rhbf, bar, out);

  (void)in_sizes; (void)n_in; (void)out_size; (void)ws_size;
}
